// DFineDecoderLayer_64544768524465
// MI455X (gfx1250) — matrix operand order repaired and checked by reading, not compiled, not run on hardware
//
#include <hip/hip_runtime.h>
#include <hip/hip_bf16.h>
#include <math.h>

// ---------------------------------------------------------------------------
// Types for CDNA5 WMMA (gfx1250, wave32)
// ---------------------------------------------------------------------------
typedef __attribute__((ext_vector_type(8)))  __bf16 v8bf;
typedef __attribute__((ext_vector_type(16))) __bf16 v16bf;
typedef __attribute__((ext_vector_type(8)))  float  v8f;
typedef __attribute__((ext_vector_type(4)))  unsigned int u32x4;
typedef __attribute__((ext_vector_type(8)))  int i32x8;
typedef __attribute__((ext_vector_type(4)))  int i32x4;

union BFrag { uint4 q[2]; v16bf v; };

#define BB 32      // batch
#define QQ 300     // queries
#define DD 256     // model dim
#define HH 8       // heads
#define HD 32      // head dim
#define FF 1024    // ffn dim
#define SS 8400    // encoder tokens
#define MROWS (BB*QQ)   // 9600
#define QP 320     // padded key/query count (divisible by 32)

#if __has_builtin(__builtin_amdgcn_tensor_load_to_lds) && __has_builtin(__builtin_amdgcn_s_wait_tensorcnt)
#define USE_TDM 1
#endif

__device__ __forceinline__ v16bf cat16(v8bf lo, v8bf hi) {
    return __builtin_shufflevector(lo, hi, 0,1,2,3,4,5,6,7,8,9,10,11,12,13,14,15);
}

#ifdef USE_TDM
// Tensor Data Mover: 1-row copy of `ndw` dwords from global to LDS.
// D# built per CDNA5 ISA 8.3/8.4: group0 = {count|flags, lds_addr,
// global_addr[31:0], global_addr[56:32]|type=2}; group1 carries data_size=4B,
// tensor_dim0 = tile_dim0 = ndw, tensor_dim1 = tile_dim1 = 1, stride = ndw.
// This toolchain exposes the 6-arg builtin (clang-23 / therock-10.0 form):
// (u32x4 g0, i32x8 g1, i32x4, i32x4, i32x8, i32 cpol)
__device__ __forceinline__ void tdm_load_b(const uint32_t* gsrc, uint32_t* lds_dst, int ndw) {
    unsigned lds_off = (unsigned)(size_t)lds_dst;
    unsigned long long ga = (unsigned long long)(size_t)gsrc;
    u32x4 g0;
    g0.x = 1u;                                                   // count=1 (valid user D#)
    g0.y = lds_off;                                              // lds_addr
    g0.z = (unsigned)(ga & 0xFFFFFFFFu);                         // global_addr[31:0]
    g0.w = (unsigned)((ga >> 32) & 0x01FFFFFFu) | (2u << 30);    // global_addr[56:32] | type=2
    unsigned nd = (unsigned)ndw;
    i32x8 g1;
    g1[0] = (int)(2u << 16);                                     // data_size = 4 bytes
    g1[1] = (int)((nd & 0xFFFFu) << 16);                         // tensor_dim0[15:0] @63:48
    g1[2] = (int)(((nd >> 16) & 0xFFFFu) | (1u << 16));          // tensor_dim0[31:16], tensor_dim1=1
    g1[3] = (int)((nd & 0xFFFFu) << 16);                         // tile_dim0 @127:112
    g1[4] = 1;                                                   // tile_dim1 = 1
    g1[5] = (int)nd;                                             // tensor_dim0_stride[31:0]
    g1[6] = 0;
    g1[7] = 0;
    i32x4 z4 = {0, 0, 0, 0};
    i32x8 z8 = {0, 0, 0, 0, 0, 0, 0, 0};
    __builtin_amdgcn_tensor_load_to_lds(g0, g1, z4, z4, z8, 0);
}
#endif

// ---------------------------------------------------------------------------
// Activation prep: bf16 copies of hidden_states and hidden+pos
// ---------------------------------------------------------------------------
__global__ void prep_act(const float* __restrict__ hs, const float* __restrict__ pos,
                         __bf16* __restrict__ hq_bf, __bf16* __restrict__ hs_bf, int n) {
    int i = blockIdx.x * blockDim.x + threadIdx.x;
    if (i < n) {
        float h = hs[i];
        hs_bf[i] = (__bf16)h;
        hq_bf[i] = (__bf16)(h + pos[i]);
    }
}

// ---------------------------------------------------------------------------
// Weight swizzle: f32 [K,N] row-major -> bf16 packed B-operand layout.
// dword index = ((nt*(K/32)+kt)*32 + lane)*8 + j
// lane<16 : K rows {2j,2j+1}; lane>=16 : K rows {16+2j,17+2j}; col = nt*16+(lane&15)
// ---------------------------------------------------------------------------
__global__ void swz_w(const float* __restrict__ W, uint32_t* __restrict__ out, int K, int N) {
    int gid = blockIdx.x * blockDim.x + threadIdx.x;
    int total = (K / 2) * N;
    if (gid >= total) return;
    int j    = gid & 7;
    int lane = (gid >> 3) & 31;
    int blk  = gid >> 8;
    int kt   = blk % (K / 32);
    int nt   = blk / (K / 32);
    int n    = nt * 16 + (lane & 15);
    int k0   = kt * 32 + (lane >> 4) * 16 + 2 * j;
    union { __bf16 h[2]; uint32_t u; } u;
    u.h[0] = (__bf16)W[(size_t)k0 * N + n];
    u.h[1] = (__bf16)W[(size_t)(k0 + 1) * N + n];
    out[gid] = u.u;
}

// ---------------------------------------------------------------------------
// WMMA GEMM: C[M,N] = A[M,K](bf16 row-major) * Bsw(K,N swizzled) + bias.
// Block: 128 threads = 4 waves; wave computes a 32x32 tile (2x2 WMMA tiles).
// grid.x = N/32, grid.y = M/128. The block's B panel (2 n-tiles x full K,
// contiguous in swizzled layout) is staged in LDS once per block via TDM
// (fallback: cooperative b128 copy), then each wave reads fragments with
// ds_load_b128. 8 loads feed 4 WMMAs per K-step.
// ---------------------------------------------------------------------------
enum { EP_F32 = 0, EP_Q, EP_K, EP_V, EP_RELU_BF, EP_SIG };

template <int FLAV>
__global__ void gemm_wmma(const __bf16* __restrict__ A, const uint32_t* __restrict__ Bsw,
                          const float* __restrict__ bias, float* __restrict__ outF,
                          __bf16* __restrict__ outBF, int M, int N, int K, float scale) {
    extern __shared__ uint32_t bsh[];
    int wave = threadIdx.x >> 5;
    int lane = threadIdx.x & 31;
    int half = lane >> 4;
    int lm   = lane & 15;
    int n0   = blockIdx.x * 32;
    int m0   = blockIdx.y * 128 + wave * 32;
    int ksteps = K >> 5;
    int ndw    = K * 16;   // dwords in the block's B panel (2 n-tiles)

    const uint32_t* gsrc = Bsw + (size_t)(blockIdx.x * 2) * ksteps * 256;
#ifdef USE_TDM
    if (wave == 0) {
        tdm_load_b(gsrc, bsh, ndw);
        __builtin_amdgcn_s_wait_tensorcnt(0);
    }
    __syncthreads();
#else
    for (int i = threadIdx.x; i < (ndw >> 2); i += 128)
        ((uint4*)bsh)[i] = ((const uint4*)gsrc)[i];
    __syncthreads();
#endif

    v8f acc[2][2] = {};
    const __bf16* arow0 = A + (size_t)(m0 + lm) * K + 16 * half;
    const __bf16* arow1 = arow0 + (size_t)16 * K;

    for (int kk = 0; kk < ksteps; ++kk) {
        v16bf a0 = cat16(*(const v8bf*)(arow0 + kk * 32),
                         *(const v8bf*)(arow0 + kk * 32 + 8));
        v16bf a1 = cat16(*(const v8bf*)(arow1 + kk * 32),
                         *(const v8bf*)(arow1 + kk * 32 + 8));
        BFrag b0, b1;
        const uint4* p0 = (const uint4*)&bsh[(size_t)kk * 256 + lane * 8];
        const uint4* p1 = (const uint4*)&bsh[(size_t)(ksteps + kk) * 256 + lane * 8];
        b0.q[0] = p0[0]; b0.q[1] = p0[1];
        b1.q[0] = p1[0]; b1.q[1] = p1[1];
        acc[0][0] = __builtin_amdgcn_wmma_f32_16x16x32_bf16(false, a0, false, b0.v, (short)0, acc[0][0], false, false);
        acc[0][1] = __builtin_amdgcn_wmma_f32_16x16x32_bf16(false, a0, false, b1.v, (short)0, acc[0][1], false, false);
        acc[1][0] = __builtin_amdgcn_wmma_f32_16x16x32_bf16(false, a1, false, b0.v, (short)0, acc[1][0], false, false);
        acc[1][1] = __builtin_amdgcn_wmma_f32_16x16x32_bf16(false, a1, false, b1.v, (short)0, acc[1][1], false, false);
    }

#pragma unroll
    for (int mi = 0; mi < 2; ++mi) {
#pragma unroll
        for (int ni = 0; ni < 2; ++ni) {
            int n = n0 + ni * 16 + lm;
            float bv = bias ? bias[n] : 0.0f;
#pragma unroll
            for (int r = 0; r < 8; ++r) {
                int m = m0 + mi * 16 + r + 8 * half;
                float v = acc[mi][ni][r] + bv;
                if (FLAV == EP_F32) {
                    outF[(size_t)m * N + n] = v;
                } else if (FLAV == EP_SIG) {
                    outF[(size_t)m * N + n] = 1.0f / (1.0f + __expf(-v));
                } else if (FLAV == EP_RELU_BF) {
                    outBF[(size_t)m * N + n] = (__bf16)fmaxf(v, 0.0f);
                } else if (FLAV == EP_Q || FLAV == EP_K) {
                    int b = m / QQ, q = m - b * QQ;
                    int h = n >> 5, d = n & 31;
                    float vv = (FLAV == EP_Q) ? v * scale : v;
                    outBF[(((size_t)(b * HH + h)) * QP + q) * HD + d] = (__bf16)vv;
                } else if (FLAV == EP_V) {
                    int b = m / QQ, q = m - b * QQ;
                    int h = n >> 5, d = n & 31;
                    outBF[(((size_t)(b * HH + h)) * HD + d) * QP + q] = (__bf16)v;
                }
            }
        }
    }
}

// ---------------------------------------------------------------------------
// Fused self-attention: per block = one (b,h) and 16 queries, 1 wave.
// scores (1 WMMA / 16 keys, K=hd=32) -> LDS, softmax (mask keys>=300),
// P*V via transposed form outT = Vt * Pt (2 m-tiles x 10 k-steps of WMMA).
// ---------------------------------------------------------------------------
__global__ void attn_kernel(const __bf16* __restrict__ qb, const __bf16* __restrict__ kb,
                            const __bf16* __restrict__ vt, __bf16* __restrict__ ctx_bf) {
    int lane = threadIdx.x & 31;
    int half = lane >> 4;
    int lm   = lane & 15;
    int qt   = blockIdx.x % 19;
    int bh   = blockIdx.x / 19;
    int b    = bh >> 3, h = bh & 7;
    int q0   = qt * 16;

    __shared__ __align__(16) float  sc[16][QP];
    __shared__ __align__(16) __bf16 pb[16][QP];

    const __bf16* qbase = qb + (size_t)bh * QP * HD;
    const __bf16* kbase = kb + (size_t)bh * QP * HD;
    const __bf16* vbase = vt + (size_t)bh * HD * QP;

    // A fragment: 16 queries x 32 dims (single K step since hd == 32)
    v8bf qlo = *(const v8bf*)(qbase + (size_t)(q0 + lm) * HD + 16 * half);
    v8bf qhi = *(const v8bf*)(qbase + (size_t)(q0 + lm) * HD + 16 * half + 8);
    v16bf aq = cat16(qlo, qhi);

    for (int kt = 0; kt < QP / 16; ++kt) {
        BFrag bf;
        const uint4* kp = (const uint4*)(kbase + (size_t)(kt * 16 + lm) * HD + 16 * half);
        bf.q[0] = kp[0];
        bf.q[1] = kp[1];
        v8f c = {};
        c = __builtin_amdgcn_wmma_f32_16x16x32_bf16(false, aq, false, bf.v,
                                                    (short)0, c, false, false);
#pragma unroll
        for (int r = 0; r < 8; ++r) sc[r + 8 * half][kt * 16 + lm] = c[r];
    }
    __syncthreads();

    // Softmax over keys < 300. Row = lm, lane halves split columns, combine via shfl.
    {
        int row = lm;
        int c0  = half * (QP / 2);
        float mx = -1e30f;
        for (int c = c0; c < c0 + QP / 2; ++c)
            if (c < QQ) mx = fmaxf(mx, sc[row][c]);
        mx = fmaxf(mx, __shfl_xor(mx, 16, 32));
        float sum = 0.0f;
        for (int c = c0; c < c0 + QP / 2; ++c) {
            float e = (c < QQ) ? __expf(sc[row][c] - mx) : 0.0f;
            sc[row][c] = e;
            sum += e;
        }
        sum += __shfl_xor(sum, 16, 32);
        float inv = 1.0f / sum;
        for (int c = c0; c < c0 + QP / 2; ++c)
            pb[row][c] = (__bf16)(sc[row][c] * inv);
    }
    __syncthreads();

    // outT[32 dims x 16 queries] = Vt[32 x 320] @ Pt[320 x 16]
    for (int mt = 0; mt < 2; ++mt) {
        v8f acc = {};
        for (int ks = 0; ks < QP / 32; ++ks) {
            const __bf16* vrow = vbase + (size_t)(mt * 16 + lm) * QP + ks * 32 + 16 * half;
            v16bf av = cat16(*(const v8bf*)vrow, *(const v8bf*)(vrow + 8));
            BFrag pf;
            const uint4* pp = (const uint4*)(&pb[lm][ks * 32 + 16 * half]);
            pf.q[0] = pp[0];
            pf.q[1] = pp[1];
            acc = __builtin_amdgcn_wmma_f32_16x16x32_bf16(false, av, false, pf.v,
                                                          (short)0, acc, false, false);
        }
        int q = q0 + lm;
        if (q < QQ) {
#pragma unroll
            for (int r = 0; r < 8; ++r) {
                int d = mt * 16 + r + 8 * half;
                ctx_bf[((size_t)(b * QQ + q)) * DD + h * HD + d] = (__bf16)acc[r];
            }
        }
    }
}

// ---------------------------------------------------------------------------
// Deformable sampling: one wave per (b,q,h), lane = channel (hd=32).
// ---------------------------------------------------------------------------
__global__ void deform_sample(const float* __restrict__ enc, const float* __restrict__ offs,
                              const float* __restrict__ awl, const float* __restrict__ refp,
                              float* __restrict__ ca) {
    int wid  = (blockIdx.x * blockDim.x + threadIdx.x) >> 5;
    int lane = threadIdx.x & 31;
    int h  = wid & 7;
    int bq = wid >> 3;          // 0 .. 9599
    int b  = bq / QQ;

    const float* rp = refp + (size_t)bq * 4;
    float cx = rp[0], cy = rp[1], rw = rp[2], rh = rp[3];
    const float* ob = offs + (size_t)bq * 192 + h * 24;
    const float* ab = awl  + (size_t)bq * 96  + h * 12;

    float aw[12];
    float mx = -1e30f;
#pragma unroll
    for (int p = 0; p < 12; ++p) { aw[p] = ab[p]; mx = fmaxf(mx, aw[p]); }
    float s = 0.0f;
#pragma unroll
    for (int p = 0; p < 12; ++p) { aw[p] = __expf(aw[p] - mx); s += aw[p]; }
    float inv = 1.0f / s;

    const int HL[3] = {80, 40, 20};
    const int WL[3] = {80, 40, 20};
    const int S0[3] = {0, 6400, 8000};

    float acc = 0.0f;
#pragma unroll
    for (int p = 0; p < 12; ++p) {
        int l = p >> 2;
        // nps = 0.25, OFFSET_SCALE = 0.5 ; grid->pixel: g = loc*W - 0.5
        float lx = cx + ob[2 * p]     * 0.25f * rw * 0.5f;
        float ly = cy + ob[2 * p + 1] * 0.25f * rh * 0.5f;
        float gx = lx * (float)WL[l] - 0.5f;
        float gy = ly * (float)HL[l] - 0.5f;
        float x0f = floorf(gx), y0f = floorf(gy);
        float wx1 = gx - x0f, wy1 = gy - y0f;
        int x0 = (int)x0f, y0 = (int)y0f;
        const float* base = enc + ((size_t)b * SS + S0[l]) * DD + h * HD + lane;
        float v = 0.0f;
        {
            int xi = x0, yi = y0; float w = (1.f - wx1) * (1.f - wy1);
            if (xi >= 0 && xi < WL[l] && yi >= 0 && yi < HL[l]) v += w * base[(size_t)(yi * WL[l] + xi) * DD];
        }
        {
            int xi = x0 + 1, yi = y0; float w = wx1 * (1.f - wy1);
            if (xi >= 0 && xi < WL[l] && yi >= 0 && yi < HL[l]) v += w * base[(size_t)(yi * WL[l] + xi) * DD];
        }
        {
            int xi = x0, yi = y0 + 1; float w = (1.f - wx1) * wy1;
            if (xi >= 0 && xi < WL[l] && yi >= 0 && yi < HL[l]) v += w * base[(size_t)(yi * WL[l] + xi) * DD];
        }
        {
            int xi = x0 + 1, yi = y0 + 1; float w = wx1 * wy1;
            if (xi >= 0 && xi < WL[l] && yi >= 0 && yi < HL[l]) v += w * base[(size_t)(yi * WL[l] + xi) * DD];
        }
        acc += aw[p] * inv * v;
    }
    ca[(size_t)bq * DD + h * HD + lane] = acc;
}

// ---------------------------------------------------------------------------
// LayerNorms (one block of 256 threads per row, D = 256)
// ---------------------------------------------------------------------------
__device__ __forceinline__ float block_sum(float v, float* red, int t) {
    red[t] = v;
    __syncthreads();
    for (int s = 128; s > 0; s >>= 1) {
        if (t < s) red[t] += red[t + s];
        __syncthreads();
    }
    float r = red[0];
    __syncthreads();
    return r;
}

__global__ void ln_add(const float* __restrict__ x, const float* __restrict__ y,
                       const float* __restrict__ w, const float* __restrict__ bb,
                       float* __restrict__ out, __bf16* __restrict__ outb) {
    __shared__ float red[256];
    int m = blockIdx.x, t = threadIdx.x;
    float v = x[(size_t)m * DD + t] + y[(size_t)m * DD + t];
    float mean = block_sum(v, red, t) * (1.0f / DD);
    float d = v - mean;
    float var = block_sum(d * d, red, t) * (1.0f / DD);
    float o = d * rsqrtf(var + 1e-5f) * w[t] + bb[t];
    out[(size_t)m * DD + t] = o;
    if (outb) outb[(size_t)m * DD + t] = (__bf16)o;
}

__global__ void gated_ln(const float* __restrict__ gates, const float* __restrict__ res,
                         const float* __restrict__ ca, const float* __restrict__ w,
                         const float* __restrict__ bb, float* __restrict__ out,
                         __bf16* __restrict__ outb) {
    __shared__ float red[256];
    int m = blockIdx.x, t = threadIdx.x;
    float g1 = gates[(size_t)m * 512 + t];
    float g2 = gates[(size_t)m * 512 + 256 + t];
    float v = g1 * res[(size_t)m * DD + t] + g2 * ca[(size_t)m * DD + t];
    float mean = block_sum(v, red, t) * (1.0f / DD);
    float d = v - mean;
    float var = block_sum(d * d, red, t) * (1.0f / DD);
    float o = d * rsqrtf(var + 1e-5f) * w[t] + bb[t];
    out[(size_t)m * DD + t] = o;
    if (outb) outb[(size_t)m * DD + t] = (__bf16)o;
}

// ---------------------------------------------------------------------------
// Pack x2 = [hid1_bf | bf16(ca)] for the gate GEMM (K = 512)
// ---------------------------------------------------------------------------
__global__ void pack_x2(const __bf16* __restrict__ hb, const float* __restrict__ ca,
                        __bf16* __restrict__ x2, int total) {
    int i = blockIdx.x * blockDim.x + threadIdx.x;
    if (i >= total) return;
    int m = i >> 9, j = i & 511;
    x2[i] = (j < 256) ? hb[(size_t)m * DD + j] : (__bf16)ca[(size_t)m * DD + (j - 256)];
}

// ---------------------------------------------------------------------------
// Host side
// ---------------------------------------------------------------------------
static inline size_t bump(size_t& off, size_t bytes) {
    size_t r = (off + 255) & ~(size_t)255;
    off = r + bytes;
    return r;
}

extern "C" void kernel_launch(void* const* d_in, const int* in_sizes, int n_in,
                              void* d_out, int out_size, void* d_ws, size_t ws_size,
                              hipStream_t stream) {
    (void)in_sizes; (void)n_in; (void)out_size; (void)ws_size;
    const float* hs   = (const float*)d_in[0];
    const float* pos  = (const float*)d_in[1];
    const float* refp = (const float*)d_in[2];
    const float* enc  = (const float*)d_in[3];
    const float* q_w  = (const float*)d_in[4];
    const float* q_b  = (const float*)d_in[5];
    const float* k_w  = (const float*)d_in[6];
    const float* k_b  = (const float*)d_in[7];
    const float* v_w  = (const float*)d_in[8];
    const float* v_b  = (const float*)d_in[9];
    const float* o_w  = (const float*)d_in[10];
    const float* o_b  = (const float*)d_in[11];
    const float* sa_ln_w = (const float*)d_in[12];
    const float* sa_ln_b = (const float*)d_in[13];
    const float* off_w   = (const float*)d_in[14];
    const float* off_b   = (const float*)d_in[15];
    const float* aw_w    = (const float*)d_in[16];
    const float* aw_b    = (const float*)d_in[17];
    const float* gate_w  = (const float*)d_in[18];
    const float* gate_b  = (const float*)d_in[19];
    const float* gate_ln_w = (const float*)d_in[20];
    const float* gate_ln_b = (const float*)d_in[21];
    const float* fc1_w = (const float*)d_in[22];
    const float* fc1_b = (const float*)d_in[23];
    const float* fc2_w = (const float*)d_in[24];
    const float* fc2_b = (const float*)d_in[25];
    const float* fin_ln_w = (const float*)d_in[26];
    const float* fin_ln_b = (const float*)d_in[27];

    char* ws = (char*)d_ws;
    size_t off = 0;
    // swizzled weights
    uint32_t* swq    = (uint32_t*)(ws + bump(off, 256 * 256 * 2));
    uint32_t* swk    = (uint32_t*)(ws + bump(off, 256 * 256 * 2));
    uint32_t* swv    = (uint32_t*)(ws + bump(off, 256 * 256 * 2));
    uint32_t* swo    = (uint32_t*)(ws + bump(off, 256 * 256 * 2));
    uint32_t* swoff  = (uint32_t*)(ws + bump(off, 256 * 192 * 2));
    uint32_t* swaw   = (uint32_t*)(ws + bump(off, 256 * 96 * 2));
    uint32_t* swgate = (uint32_t*)(ws + bump(off, 512 * 512 * 2));
    uint32_t* swfc1  = (uint32_t*)(ws + bump(off, 256 * 1024 * 2));
    uint32_t* swfc2  = (uint32_t*)(ws + bump(off, 1024 * 256 * 2));
    // activations
    __bf16* hq_bf  = (__bf16*)(ws + bump(off, (size_t)MROWS * DD * 2));
    __bf16* hs_bf  = (__bf16*)(ws + bump(off, (size_t)MROWS * DD * 2));
    __bf16* qb     = (__bf16*)(ws + bump(off, (size_t)BB * HH * QP * HD * 2));
    __bf16* kb     = (__bf16*)(ws + bump(off, (size_t)BB * HH * QP * HD * 2));
    __bf16* vt     = (__bf16*)(ws + bump(off, (size_t)BB * HH * HD * QP * 2));
    __bf16* ctx_bf = (__bf16*)(ws + bump(off, (size_t)MROWS * DD * 2));
    float*  sa     = (float*)(ws + bump(off, (size_t)MROWS * DD * 4));
    float*  hid1   = (float*)(ws + bump(off, (size_t)MROWS * DD * 4));
    __bf16* hid1_bf= (__bf16*)(ws + bump(off, (size_t)MROWS * DD * 2));
    float*  offs   = (float*)(ws + bump(off, (size_t)MROWS * 192 * 4));
    float*  awb    = (float*)(ws + bump(off, (size_t)MROWS * 96 * 4));
    float*  ca     = (float*)(ws + bump(off, (size_t)MROWS * DD * 4));
    __bf16* x2_bf  = (__bf16*)(ws + bump(off, (size_t)MROWS * 512 * 2));
    float*  gates  = (float*)(ws + bump(off, (size_t)MROWS * 512 * 4));
    float*  hid2   = (float*)(ws + bump(off, (size_t)MROWS * DD * 4));
    __bf16* hid2_bf= (__bf16*)(ws + bump(off, (size_t)MROWS * DD * 2));
    __bf16* t_bf   = (__bf16*)(ws + bump(off, (size_t)MROWS * FF * 2));
    float*  ffn    = (float*)(ws + bump(off, (size_t)MROWS * DD * 4));

    const float qscale = 1.0f / sqrtf((float)HD);
    dim3 gblk(128);

    // 1) activation prep
    {
        int n = MROWS * DD;
        prep_act<<<(n + 255) / 256, 256, 0, stream>>>(hs, pos, hq_bf, hs_bf, n);
    }
    // 2) weight swizzles
    auto swz = [&](const float* W, uint32_t* out, int K, int N) {
        int total = (K / 2) * N;
        swz_w<<<(total + 255) / 256, 256, 0, stream>>>(W, out, K, N);
    };
    swz(q_w, swq, 256, 256);   swz(k_w, swk, 256, 256);
    swz(v_w, swv, 256, 256);   swz(o_w, swo, 256, 256);
    swz(off_w, swoff, 256, 192); swz(aw_w, swaw, 256, 96);
    swz(gate_w, swgate, 512, 512);
    swz(fc1_w, swfc1, 256, 1024); swz(fc2_w, swfc2, 1024, 256);

    // 3) zero padded attention buffers (keys/queries 300..319)
    (void)hipMemsetAsync(qb, 0, (size_t)BB * HH * QP * HD * 2, stream);
    (void)hipMemsetAsync(kb, 0, (size_t)BB * HH * QP * HD * 2, stream);
    (void)hipMemsetAsync(vt, 0, (size_t)BB * HH * HD * QP * 2, stream);

    // GEMM launcher: grid (N/32, M/128), dynamic LDS = K*64 bytes (B panel)
    #define GEMM(FLAV, Abuf, Bsw_, bias_, outF_, outBF_, N_, K_, sc_) \
        gemm_wmma<FLAV><<<dim3((N_) / 32, MROWS / 128), gblk, (size_t)(K_) * 64, stream>>>( \
            Abuf, Bsw_, bias_, outF_, outBF_, MROWS, N_, K_, sc_)

    // 4) Q/K/V projections (scatter into attention layouts)
    GEMM(EP_Q, hq_bf, swq, q_b, nullptr, qb, DD, DD, qscale);
    GEMM(EP_K, hq_bf, swk, k_b, nullptr, kb, DD, DD, 1.0f);
    GEMM(EP_V, hs_bf, swv, v_b, nullptr, vt, DD, DD, 1.0f);

    // 5) fused attention
    attn_kernel<<<BB * HH * 19, 32, 0, stream>>>(qb, kb, vt, ctx_bf);

    // 6) output projection + residual LN
    GEMM(EP_F32, ctx_bf, swo, o_b, sa, nullptr, DD, DD, 1.0f);
    ln_add<<<MROWS, 256, 0, stream>>>(hs, sa, sa_ln_w, sa_ln_b, hid1, hid1_bf);

    // 7) deformable attention: offsets, weights, sampling
    GEMM(EP_F32, hid1_bf, swoff, off_b, offs, nullptr, 192, DD, 1.0f);
    GEMM(EP_F32, hid1_bf, swaw, aw_b, awb, nullptr, 96, DD, 1.0f);
    deform_sample<<<(MROWS * HH * 32) / 256, 256, 0, stream>>>(enc, offs, awb, refp, ca);

    // 8) gate
    {
        int n = MROWS * 512;
        pack_x2<<<(n + 255) / 256, 256, 0, stream>>>(hid1_bf, ca, x2_bf, n);
    }
    GEMM(EP_SIG, x2_bf, swgate, gate_b, gates, nullptr, 512, 512, 1.0f);
    gated_ln<<<MROWS, 256, 0, stream>>>(gates, hid1, ca, gate_ln_w, gate_ln_b, hid2, hid2_bf);

    // 9) FFN + final LN
    GEMM(EP_RELU_BF, hid2_bf, swfc1, fc1_b, nullptr, t_bf, FF, DD, 1.0f);
    GEMM(EP_F32, t_bf, swfc2, fc2_b, ffn, nullptr, DD, FF, 1.0f);
    ln_add<<<MROWS, 256, 0, stream>>>(hid2, ffn, fin_ln_w, fin_ln_b, (float*)d_out, nullptr);
    #undef GEMM
}
